// Llama4TextExperts_8332236554875
// MI455X (gfx1250) — compile-verified, matrix operand order checked by reading, not run on hardware
//
#include <hip/hip_runtime.h>
#include <hip/hip_bf16.h>

// ---------------------------------------------------------------------------
// MoE experts (Llama4-style): per expert e
//   gate_up = X_e @ W1_e          (1024x2048 @ 2048x8192)
//   inter   = up * silu(gate)     (split at 4096)
//   out     = inter @ W2_e        (1024x4096 @ 4096x2048)
// fp32 in/out; compute via bf16 WMMA (v_wmma_f32_16x16x32_bf16), fp32 accum.
// Block tile 64x128, 8 waves, each wave owns a 2Mx2N quad of 16x16 WMMA tiles.
// Ping-pong LDS double buffering: global->reg loads (and async global->LDS
// for the bf16 A tile in GEMM2) for step k+1 overlap the WMMAs of step k.
// ---------------------------------------------------------------------------

typedef __bf16 bf16_t;
typedef __attribute__((ext_vector_type(16))) __bf16 v16bf;
typedef __attribute__((ext_vector_type(8)))  __bf16 v8bf;
typedef __attribute__((ext_vector_type(4)))  __bf16 v4bf;
typedef __attribute__((ext_vector_type(8)))  float  v8f;
typedef __attribute__((ext_vector_type(4)))  float  v4f;
typedef __attribute__((ext_vector_type(4)))  int    v4i;

static constexpr int E      = 8;
static constexpr int T      = 1024;   // tokens per expert
static constexpr int H      = 2048;   // hidden
static constexpr int I      = 4096;   // expert dim
static constexpr int TWO_I  = 8192;   // gate_up inner dim

static constexpr int BM  = 64;        // block tile M
static constexpr int BN  = 128;       // block tile N
static constexpr int BK  = 32;        // WMMA bf16 K depth
static constexpr int LDK = 40;        // padded LDS row stride (80B, 16B-aligned rows)

// ---- gfx1250 async global->LDS path (guarded; safe if absent) ----
#if defined(__has_builtin)
#if __has_builtin(__builtin_amdgcn_global_load_async_to_lds_b128)
#define HAVE_ASYNC_LDS 1
#endif
#endif

#ifdef HAVE_ASYNC_LDS
typedef __attribute__((address_space(1))) v4i* as1_v4i_ptr;
typedef __attribute__((address_space(3))) v4i* as3_v4i_ptr;

__device__ __forceinline__ void async_copy_b128(const void* g, void* l) {
    __builtin_amdgcn_global_load_async_to_lds_b128(
        (as1_v4i_ptr)g, (as3_v4i_ptr)l, /*offset=*/0, /*cpol=*/0);
}
__device__ __forceinline__ void wait_async() {
    asm volatile("s_wait_asynccnt 0" ::: "memory");
}
#endif

__device__ __forceinline__ v8f wmma_bf16(v16bf a, v16bf b, v8f c) {
    // D = A(16x32 bf16) * B(32x16 bf16) + C(16x16 f32)
    return __builtin_amdgcn_wmma_f32_16x16x32_bf16(
        /*neg_a=*/false, a, /*neg_b=*/false, b,
        /*c_mod=*/(short)0, c, /*reuse_a=*/false, /*reuse_b=*/false);
}

// Per-lane A/B fragment from an LDS tile stored row-major, stride LDK elems.
// ISA 16-bit layout: lanes 0-15 hold K 0..7,16..23; lanes 16-31 K 8..15,24..31.
__device__ __forceinline__ v16bf load_frag(const bf16_t* p, int row, int lane) {
    const bf16_t* q = p + row * LDK + ((lane & 16) ? 8 : 0);
    v8bf lo = *(const v8bf*)(q);        // K = koff .. koff+7
    v8bf hi = *(const v8bf*)(q + 16);   // K = koff+16 .. koff+23
    v16bf r;
#pragma unroll
    for (int i = 0; i < 8; ++i) { r[i] = lo[i]; r[i + 8] = hi[i]; }
    return r;
}

// ---------------------------------------------------------------------------
// GEMM1 + SwiGLU:  inter[e,t,i] = up * silu(gate), bf16 out to workspace
// grid = (I/BN, T/BM, E), 256 threads (8 waves)
// ---------------------------------------------------------------------------
__global__ __launch_bounds__(256)
void moe_gemm1_swiglu(const float* __restrict__ X,
                      const float* __restrict__ W1,
                      bf16_t* __restrict__ inter) {
    const int e    = blockIdx.z;
    const int t0   = blockIdx.y * BM;
    const int n0   = blockIdx.x * BN;
    const int tid  = threadIdx.x;
    const int lane = tid & 31;
    const int lid  = lane & 15;
    const int w    = tid >> 5;   // wave 0..7
    const int wm2  = w & 1;      // M pair: tiles {wm2*2, wm2*2+1}
    const int wn   = w >> 1;     // N pair: tiles {wn*2,  wn*2+1}

    const float* Xe = X  + (size_t)e * T * H;
    const float* We = W1 + (size_t)e * H * TWO_I;
    bf16_t*      Ie = inter + (size_t)e * T * I;

    __shared__ alignas(16) bf16_t sA[2][BM * LDK];   // 2 x 5 KB
    __shared__ alignas(16) bf16_t sG[2][BN * LDK];   // 2 x 10 KB
    __shared__ alignas(16) bf16_t sU[2][BN * LDK];   // 2 x 10 KB

    const v8f vz = {0.f, 0.f, 0.f, 0.f, 0.f, 0.f, 0.f, 0.f};
    v8f accg[2][2] = {{vz, vz}, {vz, vz}};
    v8f accu[2][2] = {{vz, vz}, {vz, vz}};

    // staging index precompute
    const int ar0 = tid >> 3;           // A rows: ar0 and ar0+32
    const int ac0 = (tid & 7) * 4;      // A col group (float4)
    const int bn  = tid & 127;          // B: output column within tile
    const int bkb = (tid >> 7) * 16;    // B: 16 consecutive K per thread

    // register staging for next K-step
    v4f   aR0, aR1;
    float gR[16], uR[16];

    auto load_regs = [&](int kk) {
        aR0 = *(const v4f*)(Xe + (size_t)(t0 + ar0) * H + kk + ac0);
        aR1 = *(const v4f*)(Xe + (size_t)(t0 + ar0 + 32) * H + kk + ac0);
        const float* pg = We + (size_t)(kk + bkb) * TWO_I + n0 + bn;
        const float* pu = pg + I;       // up half lives at +EXPERT_DIM
#pragma unroll
        for (int j = 0; j < 16; ++j) {
            gR[j] = pg[(size_t)j * TWO_I];
            uR[j] = pu[(size_t)j * TWO_I];
        }
        if (kk + BK < H) {              // prefetch K+2 (global_prefetch_b8)
            __builtin_prefetch(pg + (size_t)BK * TWO_I, 0, 1);
            __builtin_prefetch(pu + (size_t)BK * TWO_I, 0, 1);
        }
    };

    auto store_buf = [&](int b) {
        v4bf a0, a1;
#pragma unroll
        for (int j = 0; j < 4; ++j) { a0[j] = (bf16_t)aR0[j]; a1[j] = (bf16_t)aR1[j]; }
        *(v4bf*)(&sA[b][ar0 * LDK + ac0])        = a0;
        *(v4bf*)(&sA[b][(ar0 + 32) * LDK + ac0]) = a1;
        v8bf g0, g1, u0, u1;
#pragma unroll
        for (int j = 0; j < 8; ++j) {
            g0[j] = (bf16_t)gR[j];  g1[j] = (bf16_t)gR[j + 8];
            u0[j] = (bf16_t)uR[j];  u1[j] = (bf16_t)uR[j + 8];
        }
        *(v8bf*)(&sG[b][bn * LDK + bkb])     = g0;
        *(v8bf*)(&sG[b][bn * LDK + bkb + 8]) = g1;
        *(v8bf*)(&sU[b][bn * LDK + bkb])     = u0;
        *(v8bf*)(&sU[b][bn * LDK + bkb + 8]) = u1;
    };

    // prologue: fill buffer 0
    load_regs(0);
    store_buf(0);
    __syncthreads();

    int p = 0;
    for (int k0 = 0; k0 < H; k0 += BK, p ^= 1) {
        const bool more = (k0 + BK) < H;
        if (more) load_regs(k0 + BK);   // overlap global latency with WMMAs

        // ---- compute: 2 A frags x (2 gate + 2 up) B frags -> 8 WMMAs ----
        v16bf af[2];
#pragma unroll
        for (int tm = 0; tm < 2; ++tm)
            af[tm] = load_frag(&sA[p][0], (wm2 * 2 + tm) * 16 + lid, lane);
#pragma unroll
        for (int tn = 0; tn < 2; ++tn) {
            int nt = wn * 2 + tn;
            v16bf bg = load_frag(&sG[p][0], nt * 16 + lid, lane);
            v16bf bu = load_frag(&sU[p][0], nt * 16 + lid, lane);
#pragma unroll
            for (int tm = 0; tm < 2; ++tm) {
                accg[tm][tn] = wmma_bf16(af[tm], bg, accg[tm][tn]);
                accu[tm][tn] = wmma_bf16(af[tm], bu, accu[tm][tn]);
            }
        }

        if (more) {
            store_buf(1 - p);
            __syncthreads();
        }
    }

    // ---- SwiGLU epilogue -> bf16 workspace ----
    const int mhalf = (lane & 16) ? 8 : 0;
#pragma unroll
    for (int tm = 0; tm < 2; ++tm) {
        const int row0 = t0 + (wm2 * 2 + tm) * 16 + mhalf;
#pragma unroll
        for (int tn = 0; tn < 2; ++tn) {
            const int col = n0 + (wn * 2 + tn) * 16 + lid;
#pragma unroll
            for (int r = 0; r < 8; ++r) {
                float g = accg[tm][tn][r];
                float u = accu[tm][tn][r];
                float s = g * __builtin_amdgcn_rcpf(1.0f + __expf(-g)); // silu
                Ie[(size_t)(row0 + r) * I + col] = (bf16_t)(u * s);
            }
        }
    }
}

// ---------------------------------------------------------------------------
// GEMM2:  out[e,t,h] = inter @ W2, fp32 out
// grid = (H/BN, T/BM, E), 256 threads
// ---------------------------------------------------------------------------
__global__ __launch_bounds__(256)
void moe_gemm2(const bf16_t* __restrict__ inter,
               const float* __restrict__ W2,
               float* __restrict__ Out) {
    const int e    = blockIdx.z;
    const int t0   = blockIdx.y * BM;
    const int n0   = blockIdx.x * BN;
    const int tid  = threadIdx.x;
    const int lane = tid & 31;
    const int lid  = lane & 15;
    const int w    = tid >> 5;
    const int wm2  = w & 1;
    const int wn   = w >> 1;

    const bf16_t* Ie = inter + (size_t)e * T * I;
    const float*  We = W2 + (size_t)e * I * H;

    __shared__ alignas(16) bf16_t sA[2][BM * LDK];   // 2 x 5 KB
    __shared__ alignas(16) bf16_t sB[2][BN * LDK];   // 2 x 10 KB

    const v8f vz = {0.f, 0.f, 0.f, 0.f, 0.f, 0.f, 0.f, 0.f};
    v8f acc[2][2] = {{vz, vz}, {vz, vz}};

    const int ar  = tid >> 2;           // A: row, one b128 per thread
    const int ac8 = (tid & 3) * 8;
    const int bn  = tid & 127;          // B: column
    const int bkb = (tid >> 7) * 16;

    float bR[16];

    // A tile is bf16 + layout-preserving: async DMA straight into LDS
    auto stage_A = [&](int kk, int b) {
        const bf16_t* gsrc = Ie + (size_t)(t0 + ar) * I + kk + ac8;
        bf16_t*       ldst = &sA[b][ar * LDK + ac8];
#ifdef HAVE_ASYNC_LDS
        async_copy_b128(gsrc, ldst);               // ASYNCcnt-tracked
#else
        *(v8bf*)ldst = *(const v8bf*)gsrc;
#endif
    };
    auto load_B_regs = [&](int kk) {
        const float* pb = We + (size_t)(kk + bkb) * H + n0 + bn;
#pragma unroll
        for (int j = 0; j < 16; ++j) bR[j] = pb[(size_t)j * H];
        if (kk + BK < I) __builtin_prefetch(pb + (size_t)BK * H, 0, 1);
    };
    auto store_B = [&](int b) {
        v8bf t0v, t1v;
#pragma unroll
        for (int j = 0; j < 8; ++j) { t0v[j] = (bf16_t)bR[j]; t1v[j] = (bf16_t)bR[j + 8]; }
        *(v8bf*)(&sB[b][bn * LDK + bkb])     = t0v;
        *(v8bf*)(&sB[b][bn * LDK + bkb + 8]) = t1v;
    };

    // prologue: fill buffer 0
    stage_A(0, 0);
    load_B_regs(0);
    store_B(0);
#ifdef HAVE_ASYNC_LDS
    wait_async();
#endif
    __syncthreads();

    int p = 0;
    for (int k0 = 0; k0 < I; k0 += BK, p ^= 1) {
        const bool more = (k0 + BK) < I;
        if (more) {
            stage_A(k0 + BK, 1 - p);    // async DMA overlaps WMMAs below
            load_B_regs(k0 + BK);
        }

        v16bf af[2];
#pragma unroll
        for (int tm = 0; tm < 2; ++tm)
            af[tm] = load_frag(&sA[p][0], (wm2 * 2 + tm) * 16 + lid, lane);
#pragma unroll
        for (int tn = 0; tn < 2; ++tn) {
            v16bf bf_ = load_frag(&sB[p][0], (wn * 2 + tn) * 16 + lid, lane);
#pragma unroll
            for (int tm = 0; tm < 2; ++tm)
                acc[tm][tn] = wmma_bf16(af[tm], bf_, acc[tm][tn]);
        }

        if (more) {
            store_B(1 - p);
#ifdef HAVE_ASYNC_LDS
            wait_async();               // s_wait_asynccnt 0
#endif
            __syncthreads();
        }
    }

    const int mhalf = (lane & 16) ? 8 : 0;
#pragma unroll
    for (int tm = 0; tm < 2; ++tm) {
        const size_t row0 = (size_t)e * T + t0 + (wm2 * 2 + tm) * 16 + mhalf;
#pragma unroll
        for (int tn = 0; tn < 2; ++tn) {
            const int col = n0 + (wn * 2 + tn) * 16 + lid;
#pragma unroll
            for (int r = 0; r < 8; ++r)
                Out[(row0 + r) * H + col] = acc[tm][tn][r];
        }
    }
}

// ---------------------------------------------------------------------------
extern "C" void kernel_launch(void* const* d_in, const int* in_sizes, int n_in,
                              void* d_out, int out_size, void* d_ws, size_t ws_size,
                              hipStream_t stream) {
    const float* X  = (const float*)d_in[0];   // (E*T, H)
    const float* W1 = (const float*)d_in[1];   // (E, H, 2I)
    const float* W2 = (const float*)d_in[2];   // (E, I, H)
    float* Out      = (float*)d_out;           // (E*T, H)
    bf16_t* inter   = (bf16_t*)d_ws;           // (E, T, I) bf16 = 64 MB

    dim3 g1(I / BN, T / BM, E);                // 32 x 16 x 8
    moe_gemm1_swiglu<<<g1, 256, 0, stream>>>(X, W1, inter);

    dim3 g2(H / BN, T / BM, E);                // 16 x 16 x 8
    moe_gemm2<<<g2, 256, 0, stream>>>(inter, W2, Out);
}
